// MDA_MOE_70531952935580
// MI455X (gfx1250) — matrix operand order repaired and checked by reading, not compiled, not run on hardware
//
#include <hip/hip_runtime.h>
#include <hip/hip_bf16.h>

// MDA-MOE forward for gfx1250 (MI455X). All dense GEMMs run on the matrix
// cores via v_wmma_f32_16x16x32_bf16 (fp32 accumulate); graph aggregation
// uses f32 global atomics. 32x32 output tile per wave: 4 WMMAs per K-step
// with 2x fragment reuse. Layouts follow the CDNA5 ISA 7.12.2 WMMA tables.

typedef __attribute__((ext_vector_type(16))) __bf16 bf16x16;
typedef __attribute__((ext_vector_type(8)))  float  f32x8;

__device__ __forceinline__ float eluf(float x) {
  return x > 0.f ? x : (__expf(x) - 1.f);
}

// Load one 16x32 A fragment (row-major A, row = base row + lane&15).
// Lanes 0-15 hold K{0..7,16..23}; lanes 16-31 hold K{8..15,24..31}.
__device__ __forceinline__ bf16x16 load_afrag(const float* __restrict__ A,
                                              int row, int K, int k0, int half) {
  bf16x16 a;
  const float* Ap = A + (long long)row * K + k0 + half * 8;
  #pragma unroll
  for (int e = 0; e < 8; ++e) a[e]     = (__bf16)Ap[e];
  #pragma unroll
  for (int e = 0; e < 8; ++e) a[8 + e] = (__bf16)Ap[16 + e];
  return a;
}
// Load one 32x16 B fragment (row-major B, col = base col + lane&15).
// Lanes 0-15 hold K{0..7,16..23}; lanes 16-31 hold K{8..15,24..31}.
__device__ __forceinline__ bf16x16 load_bfrag(const float* __restrict__ B,
                                              int col, int N, int k0, int half) {
  bf16x16 b;
  const float* Bp = B + (long long)(k0 + half * 8) * N + col;
  #pragma unroll
  for (int e = 0; e < 16; ++e) b[e] = (__bf16)Bp[(long long)(e + (e >= 8 ? 8 : 0)) * N];
  return b;
}
__device__ __forceinline__ void store_tile(float* __restrict__ C, const f32x8& acc,
                                           int rbase, int bcol,
                                           const float* __restrict__ bias,
                                           const float* __restrict__ addend,
                                           int N, int act) {
  const float bv = bias ? bias[bcol] : 0.f;
  #pragma unroll
  for (int v = 0; v < 8; ++v) {
    const int r = rbase + v;
    float val = acc[v] + bv;
    if (addend) val += addend[(long long)r * N + bcol];
    if (act)    val = eluf(val);
    C[(long long)r * N + bcol] = val;
  }
}

// ------------------------------------------------------------------
// GEMM: C[M,N] = act( A[M,K] @ B[K,N] + bias + addend )
// One wave per 32x32 output tile, K-step 32, bf16 WMMA, f32 accum.
// Requires M%32==0, N%32==0, K%32==0. EXEC stays full (wave-uniform exit).
// ------------------------------------------------------------------
__global__ void __launch_bounds__(128)
k_gemm_wmma(const float* __restrict__ A, const float* __restrict__ B,
            const float* __restrict__ bias, const float* __restrict__ addend,
            float* __restrict__ C, int M, int N, int K, int act)
{
  const int wave   = threadIdx.x >> 5;
  const int lane   = threadIdx.x & 31;
  const int tilesN = N >> 5;
  const int tilesT = (M >> 5) * tilesN;
  const int tile   = blockIdx.x * 4 + wave;
  if (tile >= tilesT) return;                 // wave-uniform
  const int tM = tile / tilesN, tN = tile % tilesN;
  const int half = lane >> 4, l16 = lane & 15;
  const int arow0 = tM * 32 + l16, arow1 = arow0 + 16;
  const int bcol0 = tN * 32 + l16, bcol1 = bcol0 + 16;
  f32x8 acc00 = {}, acc01 = {}, acc10 = {}, acc11 = {};
  for (int k0 = 0; k0 < K; k0 += 32) {
    const bf16x16 a0 = load_afrag(A, arow0, K, k0, half);
    const bf16x16 a1 = load_afrag(A, arow1, K, k0, half);
    const bf16x16 b0 = load_bfrag(B, bcol0, N, k0, half);
    const bf16x16 b1 = load_bfrag(B, bcol1, N, k0, half);
    acc00 = __builtin_amdgcn_wmma_f32_16x16x32_bf16(false, a0, false, b0, (short)0, acc00, false, false);
    acc01 = __builtin_amdgcn_wmma_f32_16x16x32_bf16(false, a0, false, b1, (short)0, acc01, false, false);
    acc10 = __builtin_amdgcn_wmma_f32_16x16x32_bf16(false, a1, false, b0, (short)0, acc10, false, false);
    acc11 = __builtin_amdgcn_wmma_f32_16x16x32_bf16(false, a1, false, b1, (short)0, acc11, false, false);
  }
  const int r0 = tM * 32 + half * 8;          // rows of first 16-row tile
  const int r1 = r0 + 16;                     // rows of second 16-row tile
  store_tile(C, acc00, r0, bcol0, bias, addend, N, act);
  store_tile(C, acc01, r0, bcol1, bias, addend, N, act);
  store_tile(C, acc10, r1, bcol0, bias, addend, N, act);
  store_tile(C, acc11, r1, bcol1, bias, addend, N, act);
}

// ------------------------- elementwise / graph kernels -------------------------
__global__ void k_fill(float* p, float v, long long n) {
  long long i = (long long)blockIdx.x * blockDim.x + threadIdx.x;
  if (i < n) p[i] = v;
}
__global__ void k_count(float* deg, const int* __restrict__ idx, int E) {
  int i = blockIdx.x * blockDim.x + threadIdx.x;
  if (i < E) atomicAdd(&deg[idx[i]], 1.0f);
}
__global__ void k_degprep(float* cl, float* rs, const float* __restrict__ deg, int n) {
  int i = blockIdx.x * blockDim.x + threadIdx.x;
  if (i < n) { float d = fmaxf(deg[i], 1.0f); cl[i] = d; rs[i] = rsqrtf(d); }
}
__global__ void k_rowscale(float* out, const float* __restrict__ in,
                           const float* __restrict__ s, long long total, int F) {
  long long i = (long long)blockIdx.x * blockDim.x + threadIdx.x;
  if (i < total) out[i] = in[i] * s[i / F];
}
__global__ void k_rowdiv(float* out, const float* __restrict__ in,
                         const float* __restrict__ s, long long total, int F) {
  long long i = (long long)blockIdx.x * blockDim.x + threadIdx.x;
  if (i < total) out[i] = in[i] / s[i / F];
}
__global__ void k_scatter_add(float* out, const float* __restrict__ in,
                              const int* __restrict__ src, const int* __restrict__ dst,
                              long long total, int F) {
  long long i = (long long)blockIdx.x * blockDim.x + threadIdx.x;
  if (i >= total) return;
  long long e = i / F, f = i % F;
  atomicAdd(&out[(long long)dst[e] * F + f], in[(long long)src[e] * F + f]);
}
__global__ void k_scatter_add_scaled(float* out, const float* __restrict__ z,
                                     const float* __restrict__ alpha,
                                     const int* __restrict__ src, const int* __restrict__ dst,
                                     long long total, int F) {
  long long i = (long long)blockIdx.x * blockDim.x + threadIdx.x;
  if (i >= total) return;
  long long e = i / F, f = i % F;
  atomicAdd(&out[(long long)dst[e] * F + f], alpha[e] * z[(long long)src[e] * F + f]);
}
__global__ void k_add(float* out, const float* __restrict__ a,
                      const float* __restrict__ b, long long n) {
  long long i = (long long)blockIdx.x * blockDim.x + threadIdx.x;
  if (i < n) out[i] = a[i] + b[i];
}
__global__ void k_elu(float* p, long long n) {
  long long i = (long long)blockIdx.x * blockDim.x + threadIdx.x;
  if (i < n) p[i] = eluf(p[i]);
}
__global__ void k_bias_act(float* p, const float* __restrict__ b, long long total,
                           int F, int act) {
  long long i = (long long)blockIdx.x * blockDim.x + threadIdx.x;
  if (i >= total) return;
  float v = p[i] + b[i % F];
  p[i] = act ? eluf(v) : v;
}
// GAT helpers
__global__ void k_matvec2(float* zl, float* zr, const float* __restrict__ z,
                          const float* __restrict__ al, const float* __restrict__ ar,
                          int n, int F) {
  int r = blockIdx.x * blockDim.x + threadIdx.x;
  if (r >= n) return;
  const float* row = z + (long long)r * F;
  float sl = 0.f, sr = 0.f;
  for (int f = 0; f < F; ++f) { sl += row[f] * al[f]; sr += row[f] * ar[f]; }
  zl[r] = sl; zr[r] = sr;
}
__global__ void k_edge_score(float* e, const float* __restrict__ zl,
                             const float* __restrict__ zr,
                             const int* __restrict__ src, const int* __restrict__ dst, int E) {
  int i = blockIdx.x * blockDim.x + threadIdx.x;
  if (i >= E) return;
  float v = zl[src[i]] + zr[dst[i]];
  e[i] = v > 0.f ? v : 0.2f * v;
}
__device__ __forceinline__ void atomicMaxF(float* a, float v) {
  unsigned* ua = (unsigned*)a;
  unsigned old = *ua;
  while (__uint_as_float(old) < v) {
    unsigned prev = atomicCAS(ua, old, __float_as_uint(v));
    if (prev == old) break;
    old = prev;
  }
}
__global__ void k_segmax(float* mx, const float* __restrict__ e,
                         const int* __restrict__ dst, int E) {
  int i = blockIdx.x * blockDim.x + threadIdx.x;
  if (i < E) atomicMaxF(&mx[dst[i]], e[i]);
}
__global__ void k_edge_exp(float* ex, float* den, const float* __restrict__ e,
                           const float* __restrict__ mx, const int* __restrict__ dst, int E) {
  int i = blockIdx.x * blockDim.x + threadIdx.x;
  if (i >= E) return;
  float v = __expf(e[i] - mx[dst[i]]);
  ex[i] = v;
  atomicAdd(&den[dst[i]], v);
}
__global__ void k_edge_norm(float* alpha, const float* __restrict__ ex,
                            const float* __restrict__ den, const int* __restrict__ dst, int E) {
  int i = blockIdx.x * blockDim.x + threadIdx.x;
  if (i < E) alpha[i] = ex[i] / den[dst[i]];
}
// Gate
__global__ void k_gate_base(float* base, const float* __restrict__ g1,
                            const float* __restrict__ W2, const float* __restrict__ b2, int n) {
  int r = blockIdx.x * blockDim.x + threadIdx.x;
  if (r >= n) return;
  const float* row = g1 + (long long)r * 128;
  float l[4] = {b2[0], b2[1], b2[2], b2[3]};
  for (int k = 0; k < 128; ++k) {
    float g = row[k];
    l[0] += g * W2[k * 4 + 0]; l[1] += g * W2[k * 4 + 1];
    l[2] += g * W2[k * 4 + 2]; l[3] += g * W2[k * 4 + 3];
  }
  float m = fmaxf(fmaxf(l[0], l[1]), fmaxf(l[2], l[3]));
  float s = 0.f;
  #pragma unroll
  for (int j = 0; j < 4; ++j) { l[j] = __expf(l[j] - m); s += l[j]; }
  #pragma unroll
  for (int j = 0; j < 4; ++j) base[r * 4 + j] = l[j] / s;
}
__global__ void k_quality(float* q, const float* __restrict__ o0, const float* __restrict__ o1,
                          const float* __restrict__ o2, const float* __restrict__ o3, int n) {
  int i = blockIdx.x * blockDim.x + threadIdx.x;
  if (i >= n * 4) return;
  int node = i >> 2, e = i & 3;
  const float* o = (e == 0 ? o0 : e == 1 ? o1 : e == 2 ? o2 : o3) + (long long)node * 512;
  float mean = 0.f;
  for (int f = 0; f < 512; ++f) mean += o[f];
  mean *= (1.f / 512.f);
  float var = 0.f, ent = 0.f;
  for (int f = 0; f < 512; ++f) {
    float x = o[f], d = x - mean;
    var += d * d;
    ent -= x * __logf(fmaxf(x, 0.f) + 1e-10f);
  }
  q[i] = (var / 511.f) * ent;
}
__global__ void k_gate_w(float* w, const float* __restrict__ base, const float* __restrict__ q,
                         const float* __restrict__ perf, const float* __restrict__ temp, int n) {
  int r = blockIdx.x * blockDim.x + threadIdx.x;
  if (r >= n) return;
  float t = temp[0], l[4];
  #pragma unroll
  for (int j = 0; j < 4; ++j) l[j] = base[r * 4 + j] * perf[j] * q[r * 4 + j] / t;
  float m = fmaxf(fmaxf(l[0], l[1]), fmaxf(l[2], l[3]));
  float s = 0.f;
  #pragma unroll
  for (int j = 0; j < 4; ++j) { l[j] = __expf(l[j] - m); s += l[j]; }
  #pragma unroll
  for (int j = 0; j < 4; ++j) w[r * 4 + j] = l[j] / s;
}
__global__ void k_combine(float* out, const float* __restrict__ o0, const float* __restrict__ o1,
                          const float* __restrict__ o2, const float* __restrict__ o3,
                          const float* __restrict__ w, long long total) {
  long long i = (long long)blockIdx.x * blockDim.x + threadIdx.x;
  if (i >= total) return;
  long long node = i / 512;
  out[i] = w[node * 4 + 0] * o0[i] + w[node * 4 + 1] * o1[i] +
           w[node * 4 + 2] * o2[i] + w[node * 4 + 3] * o3[i];
}
__global__ void k_accum_w(float* res, const float* __restrict__ t,
                          const float* __restrict__ w, int e, long long total) {
  long long i = (long long)blockIdx.x * blockDim.x + threadIdx.x;
  if (i >= total) return;
  res[i] += w[(i / 512) * 4 + e] * t[i];
}
__global__ void k_fuse(float* out, const float* __restrict__ a, const float* __restrict__ b,
                       const float* __restrict__ fp, long long n) {
  long long i = (long long)blockIdx.x * blockDim.x + threadIdx.x;
  if (i >= n) return;
  float f = fp[0];
  out[i] = f * a[i] + (1.f - f) * b[i];
}
__global__ void k_pairs(float* emb, const float* __restrict__ emb_mm,
                        const float* __restrict__ emb_dd, const int* __restrict__ samples,
                        long long total) {
  long long i = (long long)blockIdx.x * blockDim.x + threadIdx.x;
  if (i >= total) return;
  long long row = i / 1024; int c = (int)(i % 1024);
  if (c < 512) emb[i] = emb_mm[(long long)samples[row * 2 + 0] * 512 + c];
  else         emb[i] = emb_dd[(long long)samples[row * 2 + 1] * 512 + (c - 512)];
}
__global__ void k_head(float* result, const float* __restrict__ hidden,
                       const float* __restrict__ W1, const float* __restrict__ b1, int ns) {
  int r = blockIdx.x * blockDim.x + threadIdx.x;
  if (r >= ns) return;
  const float* row = hidden + (long long)r * 256;
  float s = b1[0];
  for (int k = 0; k < 256; ++k) s += row[k] * W1[k];
  result[r] = 1.f / (1.f + __expf(-s));
}

// ------------------------- host orchestration -------------------------
static inline unsigned gblk(long long total) { return (unsigned)((total + 255) / 256); }

static void gemm(hipStream_t st, const float* A, const float* B, const float* bias,
                 const float* add, float* C, int M, int N, int K, int act) {
  int tiles = (M / 32) * (N / 32);
  k_gemm_wmma<<<dim3((tiles + 3) / 4), dim3(128), 0, st>>>(A, B, bias, add, C, M, N, K, act);
}
static void fillk(hipStream_t st, float* p, float v, long long n) {
  k_fill<<<gblk(n), 256, 0, st>>>(p, v, n);
}

struct Ws {
  float *S1, *S2, *P0, *P1;
  float *o0, *o1, *o2, *o3;
  float *din_raw, *dout_raw, *din_cl, *ci, *co;
  float *zl, *zr, *mx, *den;
  float *ee, *ex, *alp;
  float *g1, *base, *qual, *wgt;
};

static void run_branch(hipStream_t st, const float* const* q, const float* x, int n, int inF,
                       const int* src, const int* dst, int E,
                       float* embOut, float* resOut, const Ws& w)
{
  const int d[4] = {inF, 640, 320, 512};
  // degrees + symmetric-norm coefficients
  fillk(st, w.din_raw, 0.f, n);
  fillk(st, w.dout_raw, 0.f, n);
  k_count<<<gblk(E), 256, 0, st>>>(w.din_raw, dst, E);
  k_count<<<gblk(E), 256, 0, st>>>(w.dout_raw, src, E);
  k_degprep<<<gblk(n), 256, 0, st>>>(w.din_cl, w.co, w.din_raw, n);    // co = din^-1/2
  k_degprep<<<gblk(n), 256, 0, st>>>(w.dout_raw, w.ci, w.dout_raw, n); // ci = dout^-1/2

  // ---- GCN expert -> o0
  { const float* h = x; int Kd = d[0];
    for (int l = 0; l < 3; ++l) {
      int Nl = d[l + 1];
      long long tot = (long long)n * Kd;
      k_rowscale<<<gblk(tot), 256, 0, st>>>(w.S1, h, w.ci, tot, Kd);
      fillk(st, w.S2, 0.f, tot);
      k_scatter_add<<<gblk((long long)E * Kd), 256, 0, st>>>(w.S2, w.S1, src, dst, (long long)E * Kd, Kd);
      k_rowscale<<<gblk(tot), 256, 0, st>>>(w.S2, w.S2, w.co, tot, Kd);
      float* out = (l == 2) ? w.o0 : (l == 0 ? w.P0 : w.P1);
      gemm(st, w.S2, q[2 * l], q[2 * l + 1], nullptr, out, n, Nl, Kd, l < 2 ? 1 : 0);
      h = out; Kd = Nl;
    } }
  // ---- GIN expert -> o1
  { const float* h = x; int Kd = d[0];
    for (int i = 0; i < 3; ++i) {
      int Ni = d[i + 1];
      long long tot = (long long)n * Kd;
      fillk(st, w.S1, 0.f, tot);
      k_scatter_add<<<gblk((long long)E * Kd), 256, 0, st>>>(w.S1, h, src, dst, (long long)E * Kd, Kd);
      k_add<<<gblk(tot), 256, 0, st>>>(w.S1, w.S1, h, tot);
      gemm(st, w.S1, q[8 + 4 * i], q[9 + 4 * i], nullptr, w.S2, n, Ni, Kd, 1);
      float* out = (i == 2) ? w.o1 : (i == 0 ? w.P0 : w.P1);
      gemm(st, w.S2, q[10 + 4 * i], q[11 + 4 * i], nullptr, out, n, Ni, Ni, 1);
      if (i < 2) k_elu<<<gblk((long long)n * Ni), 256, 0, st>>>(out, (long long)n * Ni);
      h = out; Kd = Ni;
    } }
  // ---- SAGE expert -> o2
  { const float* h = x; int Kd = d[0];
    for (int i = 0; i < 3; ++i) {
      int Ni = d[i + 1];
      long long tot = (long long)n * Kd;
      fillk(st, w.S1, 0.f, tot);
      k_scatter_add<<<gblk((long long)E * Kd), 256, 0, st>>>(w.S1, h, src, dst, (long long)E * Kd, Kd);
      k_rowdiv<<<gblk(tot), 256, 0, st>>>(w.S1, w.S1, w.din_cl, tot, Kd);
      gemm(st, w.S1, q[23 + 3 * i], q[24 + 3 * i], nullptr, w.S2, n, Ni, Kd, 0);
      float* out = (i == 2) ? w.o2 : (i == 0 ? w.P0 : w.P1);
      gemm(st, h, q[22 + 3 * i], nullptr, w.S2, out, n, Ni, Kd, 1);
      h = out; Kd = Ni;
    } }
  // ---- GAT expert -> o3
  { const float* h = x; int Kd = d[0];
    for (int i = 0; i < 3; ++i) {
      int Ni = d[i + 1];
      gemm(st, h, q[33 + 4 * i], nullptr, nullptr, w.S1, n, Ni, Kd, 0);  // z
      k_matvec2<<<gblk(n), 256, 0, st>>>(w.zl, w.zr, w.S1, q[34 + 4 * i], q[35 + 4 * i], n, Ni);
      k_edge_score<<<gblk(E), 256, 0, st>>>(w.ee, w.zl, w.zr, src, dst, E);
      fillk(st, w.mx, -1e30f, n);
      k_segmax<<<gblk(E), 256, 0, st>>>(w.mx, w.ee, dst, E);
      fillk(st, w.den, 0.f, n);
      k_edge_exp<<<gblk(E), 256, 0, st>>>(w.ex, w.den, w.ee, w.mx, dst, E);
      k_edge_norm<<<gblk(E), 256, 0, st>>>(w.alp, w.ex, w.den, dst, E);
      float* out = (i == 2) ? w.o3 : (i == 0 ? w.P0 : w.P1);
      fillk(st, out, 0.f, (long long)n * Ni);
      k_scatter_add_scaled<<<gblk((long long)E * Ni), 256, 0, st>>>(out, w.S1, w.alp, src, dst,
                                                                    (long long)E * Ni, Ni);
      k_bias_act<<<gblk((long long)n * Ni), 256, 0, st>>>(out, q[36 + 4 * i], (long long)n * Ni,
                                                          Ni, i < 2 ? 1 : 0);
      h = out; Kd = Ni;
    } }
  // ---- gate
  gemm(st, x, q[47], q[48], nullptr, w.g1, n, 128, inF, 1);
  k_gate_base<<<gblk(n), 256, 0, st>>>(w.base, w.g1, q[49], q[50], n);
  k_quality<<<gblk((long long)n * 4), 256, 0, st>>>(w.qual, w.o0, w.o1, w.o2, w.o3, n);
  k_gate_w<<<gblk(n), 256, 0, st>>>(w.wgt, w.base, w.qual, q[51], q[52], n);
  k_combine<<<gblk((long long)n * 512), 256, 0, st>>>(embOut, w.o0, w.o1, w.o2, w.o3, w.wgt,
                                                      (long long)n * 512);
  // ---- weighted residuals (one GEMM per expert, accumulated)
  fillk(st, resOut, 0.f, (long long)n * 512);
  const int ridx[4][2] = {{6, 7}, {20, 21}, {31, 32}, {45, 46}};
  for (int e = 0; e < 4; ++e) {
    gemm(st, x, q[ridx[e][0]], q[ridx[e][1]], nullptr, w.S1, n, 512, inF, 1);
    k_accum_w<<<gblk((long long)n * 512), 256, 0, st>>>(resOut, w.S1, w.wgt, e, (long long)n * 512);
  }
}

extern "C" void kernel_launch(void* const* d_in, const int* in_sizes, int n_in,
                              void* d_out, int out_size, void* d_ws, size_t ws_size,
                              hipStream_t stream)
{
  if (n_in < 179) return;
  const int NM = 2048, ND = 2048, NT = 4096, NS = 8192;

  // ---- parameter pointers (setup_inputs() insertion order, recursive)
  const float* lin_m = (const float*)d_in[0];
  const float* lin_d = (const float*)d_in[1];
  const float* qm[53]; const float* qd[53]; const float* qa[53];
  for (int i = 0; i < 53; ++i) {
    qm[i] = (const float*)d_in[2 + i];
    qd[i] = (const float*)d_in[55 + i];
    qa[i] = (const float*)d_in[108 + i];
  }
  const float* mlp0W = (const float*)d_in[161];
  const float* mlp0b = (const float*)d_in[162];
  const float* mlp1W = (const float*)d_in[163];
  const float* mlp1b = (const float*)d_in[164];
  const float* f_m  = (const float*)d_in[165];
  const float* f_d  = (const float*)d_in[166];
  const float* f_md = (const float*)d_in[167];
  const float* f_w1 = (const float*)d_in[168];
  const float* f_w2 = (const float*)d_in[169];
  const float* miRNA   = (const float*)d_in[170];
  const float* disease = (const float*)d_in[171];
  const int* mm_src = (const int*)d_in[172];
  const int* mm_dst = (const int*)d_in[173];
  const int* dd_src = (const int*)d_in[174];
  const int* dd_dst = (const int*)d_in[175];
  const int* md_src = (const int*)d_in[176];
  const int* md_dst = (const int*)d_in[177];
  const int* samples = (const int*)d_in[178];
  const int E_mm = in_sizes[172], E_dd = in_sizes[174], E_md = in_sizes[176];
  const int E_max = E_md > E_mm ? (E_md > E_dd ? E_md : E_dd) : (E_mm > E_dd ? E_mm : E_dd);

  // ---- workspace bump allocator
  char* wp = (char*)d_ws;
  auto alloc = [&](long long elems) -> float* {
    float* p = (float*)wp;
    wp += ((size_t)elems * 4 + 255) & ~(size_t)255;
    return p;
  };
  const long long BIG = 4ll * 1024 * 1024;   // max n*F elements (2048*2048 / 4096*1024)
  Ws w;
  w.S1 = alloc(BIG); w.S2 = alloc(BIG); w.P0 = alloc(BIG); w.P1 = alloc(BIG);
  w.o0 = alloc((long long)NT * 512); w.o1 = alloc((long long)NT * 512);
  w.o2 = alloc((long long)NT * 512); w.o3 = alloc((long long)NT * 512);
  w.din_raw = alloc(NT); w.dout_raw = alloc(NT); w.din_cl = alloc(NT);
  w.ci = alloc(NT); w.co = alloc(NT);
  w.zl = alloc(NT); w.zr = alloc(NT); w.mx = alloc(NT); w.den = alloc(NT);
  w.ee = alloc(E_max); w.ex = alloc(E_max); w.alp = alloc(E_max);
  w.g1 = alloc((long long)NT * 128);
  w.base = alloc(NT * 4); w.qual = alloc(NT * 4); w.wgt = alloc(NT * 4);
  float* mdbuf = alloc((long long)NT * 1024);
  float* embM = alloc((long long)NM * 512); float* resM = alloc((long long)NM * 512);
  float* embD = alloc((long long)ND * 512); float* resD = alloc((long long)ND * 512);
  float* embA = alloc((long long)NT * 512); float* resA = alloc((long long)NT * 512);
  float* emb_mm = alloc((long long)NM * 512); float* emb_dd = alloc((long long)ND * 512);
  float* hidden = alloc((long long)NS * 256);
  if ((size_t)(wp - (char*)d_ws) > ws_size) return;  // ws too small: do nothing

  // ---- output layout: result | emb_m | emb_mm_ass | emb_d | emb_dd_ass | emb
  float* out = (float*)d_out;
  float* o_result = out;
  float* o_embm   = out + NS;
  float* o_mmass  = o_embm + (long long)NM * 512;
  float* o_embd   = o_mmass + (long long)NM * 512;
  float* o_ddass  = o_embd + (long long)ND * 512;
  float* o_emb    = o_ddass + (long long)ND * 512;

  // ---- branches m, d
  run_branch(stream, qm, miRNA,   NM, 2048, mm_src, mm_dst, E_mm, embM, resM, w);
  run_branch(stream, qd, disease, ND, 2048, dd_src, dd_dst, E_dd, embD, resD, w);

  // ---- md features: concat(miRNA @ lin_m, disease @ lin_d)
  gemm(stream, miRNA,   lin_m, nullptr, nullptr, mdbuf,                        NM, 1024, 2048, 0);
  gemm(stream, disease, lin_d, nullptr, nullptr, mdbuf + (long long)NM * 1024, ND, 1024, 2048, 0);
  run_branch(stream, qa, mdbuf, NT, 1024, md_src, md_dst, E_md, embA, resA, w);

  // ---- fuse into outputs
  k_fuse<<<gblk((long long)NM * 512), 256, 0, stream>>>(o_embm, embM, resM, f_m, (long long)NM * 512);
  k_fuse<<<gblk((long long)ND * 512), 256, 0, stream>>>(o_embd, embD, resD, f_d, (long long)ND * 512);
  k_fuse<<<gblk((long long)NM * 512), 256, 0, stream>>>(o_mmass, embA, resA, f_md, (long long)NM * 512);
  k_fuse<<<gblk((long long)ND * 512), 256, 0, stream>>>(o_ddass, embA + (long long)NM * 512,
                                                        resA + (long long)NM * 512, f_md,
                                                        (long long)ND * 512);
  k_fuse<<<gblk((long long)NM * 512), 256, 0, stream>>>(emb_mm, o_embm, o_mmass, f_w1, (long long)NM * 512);
  k_fuse<<<gblk((long long)ND * 512), 256, 0, stream>>>(emb_dd, o_embd, o_ddass, f_w2, (long long)ND * 512);

  // ---- pair embeddings + head MLP
  k_pairs<<<gblk((long long)NS * 1024), 256, 0, stream>>>(o_emb, emb_mm, emb_dd, samples,
                                                          (long long)NS * 1024);
  gemm(stream, o_emb, mlp0W, mlp0b, nullptr, hidden, NS, 256, 1024, 1);
  k_head<<<gblk(NS), 256, 0, stream>>>(o_result, hidden, mlp1W, mlp1b, NS);
}
